// PathEncoder_44401371906318
// MI455X (gfx1250) — matrix operand order repaired and checked by reading, not compiled, not run on hardware
//
#include <hip/hip_runtime.h>
#include <hip/hip_bf16.h>
#include <math.h>

// ---------------- problem constants ----------------
#define BB      8
#define N_NODE  512
#define N_TOK   8
#define N_PATHS 256
#define PATH_LEN 64
#define HH      256
#define H2      512
#define G3      768
#define NNP     513           // N_NODE + pad row
#define XROWS   (BB * NNP)    // 4104
#define NSEQ_P  (BB * N_PATHS)// 2048
#define NXROWS  (NSEQ_P * PATH_LEN) // 131072
#define RKS     528           // padded K-stride for w_rank f16 (32B aligned)

typedef __attribute__((ext_vector_type(16))) _Float16 v16h;
typedef __attribute__((ext_vector_type(8)))  _Float16 v8h;
typedef __attribute__((ext_vector_type(8)))  float    v8f;
typedef __attribute__((ext_vector_type(4)))  int      v4i;

#ifndef __has_builtin
#define __has_builtin(x) 0
#endif
#if __has_builtin(__builtin_amdgcn_global_load_async_to_lds_b128)
#define HAVE_ASYNC_LDS 1
#else
#define HAVE_ASYNC_LDS 0
#endif

#if HAVE_ASYNC_LDS
typedef __attribute__((address_space(1))) v4i GV4;    // global int4
typedef __attribute__((address_space(3))) v4i LV4;    // LDS int4
#if __has_builtin(__builtin_amdgcn_s_wait_asynccnt)
#define AWAIT(n) __builtin_amdgcn_s_wait_asynccnt(n)
#else
#define AWAIT(n) asm volatile("s_wait_asynccnt %0" :: "i"(n) : "memory")
#endif
#endif

// ---------------- WMMA fragment helpers ----------------
// A 16x32 f16 fragment: lane m = lane&15; lanes<16 cover K {0..7,16..23},
// lanes>=16 cover K {8..15,24..31}  (per ISA 16-bit A layout table).
__device__ __forceinline__ v16h ld16h(const _Float16* p0, const _Float16* p1) {
  v8h a = *(const v8h*)p0;
  v8h b = *(const v8h*)p1;
  v16h r;
#pragma unroll
  for (int i = 0; i < 8; ++i) { r[i] = a[i]; r[i + 8] = b[i]; }
  return r;
}
__device__ __forceinline__ v16h frag_a(const _Float16* arow, int k0, int hi) {
  const _Float16* p = arow + k0 + hi * 8;
  return ld16h(p, p + 16);
}
// B 32x16 f16 fragment: lane n = lane&15; lanes<16 cover K {0..7,16..23},
// lanes>=16 cover K {8..15,24..31} (row of weight[N,K] is contiguous in K).
__device__ __forceinline__ v16h frag_b(const _Float16* brow, int k0, int hi) {
  const _Float16* p = brow + k0 + hi * 8;
  return ld16h(p, p + 16);
}

__device__ __forceinline__ void atomicMaxF32(float* addr, float val) {
  if (val >= 0.0f) atomicMax((int*)addr, __float_as_int(val));
  else             atomicMin((unsigned int*)addr, __float_as_uint(val));
}

// ---------------- small utility kernels ----------------
__global__ void cvt_f32_f16(const float* __restrict__ src, _Float16* __restrict__ dst,
                            int rows, int scols, int dcols) {
  int i = blockIdx.x * blockDim.x + threadIdx.x;
  int total = rows * dcols;
  if (i >= total) return;
  int r = i / dcols, c = i - r * dcols;
  dst[i] = (c < scols) ? (_Float16)src[(size_t)r * scols + c] : (_Float16)0.0f;
}

__global__ void fill_f32(float* p, float v, int n) {
  int i = blockIdx.x * blockDim.x + threadIdx.x;
  if (i < n) p[i] = v;
}

// x = max over tokens of emb[nodes]; pad row (node id N_NODE) = 0. Output f16.
__global__ void embed_maxpool(const int* __restrict__ nodes,
                              const float* __restrict__ emb,
                              _Float16* __restrict__ xh) {
  int blk = blockIdx.x;            // 0 .. B*NNP-1
  int b = blk / NNP, j = blk - b * NNP;
  int c = threadIdx.x;             // 0..255
  float m;
  if (j == N_NODE) {
    m = 0.0f;
  } else {
    m = -INFINITY;
    const int* np = nodes + ((size_t)b * N_NODE + j) * N_TOK;
#pragma unroll
    for (int t = 0; t < N_TOK; ++t) {
      int id = np[t];
      if (t + 1 < N_TOK)
        __builtin_prefetch(emb + (size_t)np[t + 1] * HH + c, 0, 0); // global_prefetch_b8
      m = fmaxf(m, emb[(size_t)id * HH + c]);
    }
  }
  xh[(size_t)blk * HH + c] = (_Float16)m;
}

// ---------------- generic WMMA GEMM:  C[M,N] = A[M,K](f16) * Bw[N,K]^T + bias ----------------
__global__ __launch_bounds__(256)
void wmma_gemm_bias(const _Float16* __restrict__ A, const _Float16* __restrict__ Bw,
                    const float* __restrict__ bias, float* __restrict__ C,
                    int M, int N, int K, int ldb) {
  int lane = threadIdx.x & 31, wave = threadIdx.x >> 5;
  int col0 = (blockIdx.x * 8 + wave) * 16;
  if (col0 >= N) return;
  int row0 = blockIdx.y * 16;
  int hi = lane >> 4, nl = lane & 15;
  int r = row0 + nl; if (r > M - 1) r = M - 1;     // clamp; pad rows never stored
  const _Float16* arow = A + (size_t)r * K;
  const _Float16* brow = Bw + (size_t)(col0 + nl) * ldb;
  v8f acc = {};
  for (int k0 = 0; k0 < K; k0 += 32) {
    v16h a = frag_a(arow, k0, hi);
    v16h b = frag_b(brow, k0, hi);
    acc = __builtin_amdgcn_wmma_f32_16x16x32_f16(false, a, false, b, (short)0, acc, false, false);
  }
  float bv = bias ? bias[col0 + nl] : 0.0f;
#pragma unroll
  for (int e = 0; e < 8; ++e) {
    int m = row0 + e + 8 * hi;
    if (m < M) C[(size_t)m * N + (col0 + nl)] = acc[e] + bv;
  }
}

// ---------------- GRU recurrent scan (fwd/bwd, base/path) ----------------
// Block = 16 sequences, 512 threads (16 waves). Each wave owns 3 N-tiles of the
// 768-wide gate GEMM; w_hh fragments are preloaded once into VGPRs and reused
// across every timestep. h lives in LDS (f16) and feeds WMMA A-fragments via ds.
// The next timestep's input-gate rows (xw, gathered via paths) are prefetched
// into a double-buffered LDS staging area with async global->LDS loads so the
// gather latency hides behind the WMMA gate GEMM.
__global__ __launch_bounds__(512, 1)
void gru_scan(const _Float16* __restrict__ whh, const float* __restrict__ bhh,
              const float* __restrict__ xw, const int* __restrict__ paths,
              float* __restrict__ out32, _Float16* __restrict__ out16,
              int num_seq, int T, int dir, int mode) {
  __shared__ _Float16 hbuf[16 * HH];    // 8 KB
  __shared__ float    ghbuf[16 * G3];   // 48 KB
#if HAVE_ASYNC_LDS
  __shared__ float    xwbuf[2][16 * G3]; // 96 KB double-buffered gate-input staging
#endif
  int tid = threadIdx.x, lane = tid & 31, wave = tid >> 5;
  int hi = lane >> 4, nl = lane & 15;
  int s0 = blockIdx.x * 16;

  auto rowof = [&](int m, int pos) -> int {
    int s = s0 + m;
    bool valid = s < num_seq;
    if (mode == 0) return (valid ? s : 0) * NNP + pos;     // base: seq index == batch b
    int node = valid ? paths[(size_t)s * PATH_LEN + pos] : 0;
    return valid ? ((s >> 8) * NNP + node) : 0;            // path: gather via node id
  };

#if HAVE_ASYNC_LDS
  auto prefetch_step = [&](int t, int buf) {
    int pos = dir ? (T - 1 - t) : t;
#pragma unroll
    for (int i = 0; i < 6; ++i) {
      int q = tid + i * 512;                // 3072 chunks of 16B cover 16x768 f32
      int m = q / 192, cr = q - m * 192;
      const float* g = xw + (size_t)rowof(m, pos) * G3 + cr * 4;
      __builtin_amdgcn_global_load_async_to_lds_b128(
          (GV4*)g, (LV4*)&xwbuf[buf][m * G3 + cr * 4], 0, 0);
    }
  };
  prefetch_step(0, 0);
#endif

  // preload B fragments for this wave's 3 N-tiles (held in VGPRs for the whole scan)
  v16h Bf[3][8];
#pragma unroll
  for (int j = 0; j < 3; ++j) {
    const _Float16* bp = whh + (size_t)((wave * 3 + j) * 16 + nl) * HH;
#pragma unroll
    for (int kk = 0; kk < 8; ++kk) Bf[j][kk] = frag_b(bp, kk * 32, hi);
  }

  for (int i = tid; i < 16 * HH; i += 512) hbuf[i] = (_Float16)0.0f;
  __syncthreads();

  for (int t = 0; t < T; ++t) {
    int pos = dir ? (T - 1 - t) : t;
#if HAVE_ASYNC_LDS
    if (t + 1 < T) prefetch_step(t + 1, (t + 1) & 1);      // overlap with gate GEMM
#endif
    // ---- gate GEMM: gh[16,768] = h[16,256] @ whh^T + bhh ----
#pragma unroll
    for (int j = 0; j < 3; ++j) {
      v8f acc = {};
#pragma unroll
      for (int kk = 0; kk < 8; ++kk) {
        v16h a = frag_a(hbuf + nl * HH, kk * 32, hi);      // ds reads
        acc = __builtin_amdgcn_wmma_f32_16x16x32_f16(false, a, false, Bf[j][kk],
                                                     (short)0, acc, false, false);
      }
      int n = (wave * 3 + j) * 16 + nl;
      float bb = bhh[n];
#pragma unroll
      for (int e = 0; e < 8; ++e) ghbuf[(e + 8 * hi) * G3 + n] = acc[e] + bb;
    }
#if HAVE_ASYNC_LDS
    // 6 async ops/wave issued for step t+1; waiting <=6 drains step t's batch
    if (t + 1 < T) { AWAIT(6); } else { AWAIT(0); }
#endif
    __syncthreads();
    // ---- elementwise GRU update: 16 seq x 256 ch over 512 threads ----
#pragma unroll
    for (int ii = 0; ii < 8; ++ii) {
      int idx = tid + ii * 512;
      int m = idx >> 8, c = idx & 255;
      int s = s0 + m;
      bool valid = s < num_seq;
#if HAVE_ASYNC_LDS
      const float* xr = &xwbuf[t & 1][m * G3];
#else
      const float* xr = xw + (size_t)rowof(m, pos) * G3;
#endif
      float gr  = xr[c]        + ghbuf[m * G3 + c];
      float gz  = xr[HH + c]   + ghbuf[m * G3 + HH + c];
      float gnx = xr[2*HH + c];
      float gnh = ghbuf[m * G3 + 2*HH + c];
      float rr = 1.0f / (1.0f + __expf(-gr));
      float zz = 1.0f / (1.0f + __expf(-gz));
      float nn = tanhf(gnx + rr * gnh);
      float hold = (float)hbuf[m * HH + c];
      float hnew = (1.0f - zz) * nn + zz * hold;
      hbuf[m * HH + c] = (_Float16)hnew;
      if (valid) {
        if (mode == 0) out32[((size_t)s * N_NODE + pos) * H2 + dir * HH + c] = hnew;
        else           out16[((size_t)s * PATH_LEN + pos) * H2 + dir * HH + c] = (_Float16)hnew;
      }
    }
    __syncthreads();
  }
}

// ---------------- rank projection GEMM with fused scatter-max ----------------
// out[m,n] = nx[m,:512] @ w_rank[n,:512]^T + b_rank[n] + w_rank[n,512]/(p+1)
// then atomic-max into vbuf[b, node(m), n].
__global__ __launch_bounds__(256)
void rank_gemm_scatter(const _Float16* __restrict__ A, const _Float16* __restrict__ Bw,
                       const float* __restrict__ w_rank, const float* __restrict__ b_rank,
                       const int* __restrict__ paths, float* __restrict__ vbuf) {
  int lane = threadIdx.x & 31, wave = threadIdx.x >> 5;
  int col0 = (blockIdx.x * 8 + wave) * 16;
  int row0 = blockIdx.y * 16;
  int hi = lane >> 4, nl = lane & 15;
  const _Float16* arow = A + (size_t)(row0 + nl) * H2;
  const _Float16* brow = Bw + (size_t)(col0 + nl) * RKS;
  v8f acc = {};
  for (int k0 = 0; k0 < H2; k0 += 32) {
    v16h a = frag_a(arow, k0, hi);
    v16h b = frag_b(brow, k0, hi);
    acc = __builtin_amdgcn_wmma_f32_16x16x32_f16(false, a, false, b, (short)0, acc, false, false);
  }
  int n = col0 + nl;
  float bv = b_rank[n];
  float wl = w_rank[(size_t)n * (H2 + 1) + H2];            // rank-feature column (f32)
#pragma unroll
  for (int e = 0; e < 8; ++e) {
    int m = row0 + e + 8 * hi;                             // m = ((b*256+p)*64 + t)
    int p = (m >> 6) & (N_PATHS - 1);
    float val = acc[e] + bv + wl * (1.0f / (float)(p + 1));
    int node = paths[m];
    int b = m >> 14;
    atomicMaxF32(vbuf + ((size_t)(b * NNP + node) * H2 + n), val);
  }
}

// ---------------- finalize: all_v = base_v + (scatter-max | fallback) ----------------
__global__ void finalize_add(const float* __restrict__ vbuf, const float* __restrict__ fb,
                             float* __restrict__ allv) {
  int i = blockIdx.x * blockDim.x + threadIdx.x;           // < B*N_NODE*H2
  int c = i & (H2 - 1);
  int node = (i >> 9) & (N_NODE - 1);
  int b = i >> 18;
  size_t r = (size_t)(b * NNP + node) * H2 + c;
  float vv = vbuf[r];
  allv[i] += (vv == -INFINITY) ? fb[r] : vv;
}

__global__ void colmax(const float* __restrict__ allv, float* __restrict__ out) {
  int b = blockIdx.x, c = threadIdx.x;                     // 512 threads
  float m = -INFINITY;
  for (int nd = 0; nd < N_NODE; ++nd)
    m = fmaxf(m, allv[((size_t)b * N_NODE + nd) * H2 + c]);
  out[b * H2 + c] = m;
}

__global__ void hproj(const float* __restrict__ mx, const float* __restrict__ w_h,
                      const float* __restrict__ b_h, float* __restrict__ out) {
  int b = blockIdx.x, o = threadIdx.x;                     // 256 threads
  float acc = b_h[o];
  for (int k = 0; k < H2; ++k) acc += mx[b * H2 + k] * w_h[(size_t)o * H2 + k];
  out[b * HH + o] = acc;
}

// ---------------- host launcher ----------------
extern "C" void kernel_launch(void* const* d_in, const int* in_sizes, int n_in,
                              void* d_out, int out_size, void* d_ws, size_t ws_size,
                              hipStream_t stream) {
  const int*   nodes     = (const int*)d_in[0];
  const int*   paths     = (const int*)d_in[1];
  const float* emb       = (const float*)d_in[2];
  const float* wf_ih     = (const float*)d_in[3];
  const float* wf_hh     = (const float*)d_in[4];
  const float* bf_ih     = (const float*)d_in[5];
  const float* bf_hh     = (const float*)d_in[6];
  const float* wb_ih     = (const float*)d_in[7];
  const float* wb_hh     = (const float*)d_in[8];
  const float* bb_ih     = (const float*)d_in[9];
  const float* bb_hh     = (const float*)d_in[10];
  const float* w_transit = (const float*)d_in[11];
  const float* b_transit = (const float*)d_in[12];
  const float* w_rank    = (const float*)d_in[13];
  const float* b_rank    = (const float*)d_in[14];
  const float* w_h       = (const float*)d_in[15];
  const float* b_h       = (const float*)d_in[16];

  float* allv = (float*)d_out;                       // [B, N_NODE, 2H]
  float* last = allv + (size_t)BB * N_NODE * H2;     // [B, H]

  char* ws = (char*)d_ws;
  auto alloc = [&](size_t bytes) -> char* {
    char* p = ws; ws += (bytes + 255) & ~(size_t)255; return p;
  };
  _Float16* x_h     = (_Float16*)alloc((size_t)XROWS * HH * 2);
  float*    xw_f    = (float*)   alloc((size_t)XROWS * G3 * 4);
  float*    xw_b    = (float*)   alloc((size_t)XROWS * G3 * 4);
  _Float16* wfih_h  = (_Float16*)alloc((size_t)G3 * HH * 2);
  _Float16* wbih_h  = (_Float16*)alloc((size_t)G3 * HH * 2);
  _Float16* wfhh_h  = (_Float16*)alloc((size_t)G3 * HH * 2);
  _Float16* wbhh_h  = (_Float16*)alloc((size_t)G3 * HH * 2);
  _Float16* wtr_h   = (_Float16*)alloc((size_t)H2 * HH * 2);
  _Float16* wrank_h = (_Float16*)alloc((size_t)H2 * RKS * 2);
  _Float16* nx_h    = (_Float16*)alloc((size_t)NXROWS * H2 * 2);
  float*    vbuf    = (float*)   alloc((size_t)BB * NNP * H2 * 4);
  float*    fb      = (float*)   alloc((size_t)XROWS * H2 * 4);
  float*    avmax   = (float*)   alloc((size_t)BB * H2 * 4);

  auto cvt = [&](const float* s, _Float16* d, int rows, int sc, int dc) {
    int total = rows * dc;
    hipLaunchKernelGGL(cvt_f32_f16, dim3((total + 255) / 256), dim3(256), 0, stream,
                       s, d, rows, sc, dc);
  };
  cvt(wf_ih, wfih_h, G3, HH, HH);
  cvt(wb_ih, wbih_h, G3, HH, HH);
  cvt(wf_hh, wfhh_h, G3, HH, HH);
  cvt(wb_hh, wbhh_h, G3, HH, HH);
  cvt(w_transit, wtr_h, H2, HH, HH);
  cvt(w_rank, wrank_h, H2, H2 + 1, RKS);

  hipLaunchKernelGGL(embed_maxpool, dim3(BB * NNP), dim3(HH), 0, stream, nodes, emb, x_h);

  // input-side gates for both directions (pad row -> b_ih automatically)
  hipLaunchKernelGGL(wmma_gemm_bias, dim3(G3 / 128, (XROWS + 15) / 16), dim3(256), 0, stream,
                     x_h, wfih_h, bf_ih, xw_f, XROWS, G3, HH, HH);
  hipLaunchKernelGGL(wmma_gemm_bias, dim3(G3 / 128, (XROWS + 15) / 16), dim3(256), 0, stream,
                     x_h, wbih_h, bb_ih, xw_b, XROWS, G3, HH, HH);

  // base bidirectional GRU -> writes base_v directly into d_out halves
  hipLaunchKernelGGL(gru_scan, dim3(1), dim3(512), 0, stream,
                     wfhh_h, bf_hh, xw_f, (const int*)nullptr, allv, (_Float16*)nullptr,
                     BB, N_NODE, 0, 0);
  hipLaunchKernelGGL(gru_scan, dim3(1), dim3(512), 0, stream,
                     wbhh_h, bb_hh, xw_b, (const int*)nullptr, allv, (_Float16*)nullptr,
                     BB, N_NODE, 1, 0);
  // path bidirectional GRU -> nx (f16)
  hipLaunchKernelGGL(gru_scan, dim3(NSEQ_P / 16), dim3(512), 0, stream,
                     wfhh_h, bf_hh, xw_f, paths, (float*)nullptr, nx_h,
                     NSEQ_P, PATH_LEN, 0, 1);
  hipLaunchKernelGGL(gru_scan, dim3(NSEQ_P / 16), dim3(512), 0, stream,
                     wbhh_h, bb_hh, xw_b, paths, (float*)nullptr, nx_h,
                     NSEQ_P, PATH_LEN, 1, 1);

  // scatter-max pool (vbuf init to -inf, then fused rank-proj + atomic max)
  int vn = BB * NNP * H2;
  hipLaunchKernelGGL(fill_f32, dim3((vn + 255) / 256), dim3(256), 0, stream,
                     vbuf, -INFINITY, vn);
  hipLaunchKernelGGL(rank_gemm_scatter, dim3(H2 / 128, NXROWS / 16), dim3(256), 0, stream,
                     nx_h, wrank_h, w_rank, b_rank, paths, vbuf);

  // fallback = x @ w_transit^T + b_transit
  hipLaunchKernelGGL(wmma_gemm_bias, dim3(H2 / 128, (XROWS + 15) / 16), dim3(256), 0, stream,
                     x_h, wtr_h, b_transit, fb, XROWS, H2, HH, HH);

  int an = BB * N_NODE * H2;
  hipLaunchKernelGGL(finalize_add, dim3(an / 256), dim3(256), 0, stream, vbuf, fb, allv);
  hipLaunchKernelGGL(colmax, dim3(BB), dim3(H2), 0, stream, allv, avmax);
  hipLaunchKernelGGL(hproj, dim3(BB), dim3(HH), 0, stream, avmax, w_h, b_h, last);
}
